// Linker_52707838656531
// MI455X (gfx1250) — matrix operand order repaired and checked by reading, not compiled, not run on hardware
//
#include <hip/hip_runtime.h>
#include <hip/hip_bf16.h>

// ---------------------------------------------------------------------------
// Flow-chaining trajectory linker for MI455X (gfx1250, wave32).
//
// Problem constants (from the reference setup):
#define HH      480
#define WW      1024
#define SS      32
#define HW      (HH*WW)           // 491520
#define NSLOTS  (2*HW/16)         // 61440  (sample_step=4)
#define MARGIN  10
#define GXN     252               // arange(10,1015,4)
#define GYN     116               // arange(10, 471,4)
#define GTOT    (GXN*GYN)         // 29232
#define TI      (HH/16)           // 30 row tiles
#define TJ      (WW/16)           // 64 col tiles
//
// Per-step pipeline (stream-ordered, 31 dependent steps):
//   prune -> advect -> hpassWMMA -> compact(slot scan + births)
// Dilation = separable 5-tap ones-conv. Horizontal pass: two exact f16 WMMAs
// per 16x16 tile (banded matmul; occ in {0,1}, partial sums <= 5 exact, f32
// accum). Vertical 5-sum evaluated only at the G grid points inside compact.
// ---------------------------------------------------------------------------

typedef __attribute__((ext_vector_type(16))) _Float16 v16h;
typedef __attribute__((ext_vector_type(8)))  _Float16 v8h;
typedef __attribute__((ext_vector_type(8)))  float    v8f;

// ---- constant B operands (band matrices), baked at compile time -----------
// B layout (32x16 f16): lane N=lane&15, bank=lane>>4; halves h<8 carry
// K=bank*8+h, halves h>=8 carry K=16+bank*8+(h-8). Packed as 8 dwords (2 halves each).
//   mat 0: [left|center] K=0..31 maps to source cols 16(J-1)+K,
//          value = (|n + 16 - K| <= 2)
//   mat 1: right halo, K'=0..15 maps to source cols 16(J+1)+K',
//          value = (|n - 16 - K'| <= 2)   (upper K unused: A2 zero-padded)
struct alignas(32) BTabT { unsigned v[2][32][8]; };

constexpr BTabT makeBTab() {
  BTabT t{};
  for (int mat = 0; mat < 2; ++mat)
    for (int lane = 0; lane < 32; ++lane)
      for (int d = 0; d < 8; ++d) {
        unsigned w = 0;
        int n = lane & 15;
        for (int part = 0; part < 2; ++part) {
          int h = 2 * d + part;
          int K = (h < 8) ? ((lane >= 16) ? 8 : 0) + h : 16 + ((lane >= 16) ? 8 : 0) + (h - 8);
          int delta = (mat == 0) ? (n + 16 - K) : (n - 16 - K);
          bool one = (delta >= -2 && delta <= 2);
          if (mat == 1 && lane >= 16) one = false;  // A2 upper-K is zero
          if (one) w |= (part ? 0x3C000000u : 0x00003C00u);  // 1.0h
        }
        t.v[mat][lane][d] = w;
      }
  return t;
}
static __device__ __constant__ BTabT kB = makeBTab();

// ---- bilinear sampling with reference semantics ---------------------------
// weights from UNclipped floor coords, gather indices clipped to the image.
struct Bil { int x0, x1, y0, y1; float w00, w01, w10, w11; };

__device__ __forceinline__ Bil mk_bil(float x, float y) {
  float fx = floorf(x), fy = floorf(y);
  float wx1 = x - fx, wx0 = 1.0f - wx1;
  float wy1 = y - fy, wy0 = 1.0f - wy1;
  // clamp in float first to keep the int cast defined, == jnp.clip(int32(...))
  float fxc = fminf(fmaxf(fx, 0.0f), (float)(WW - 1));
  float fyc = fminf(fmaxf(fy, 0.0f), (float)(HH - 1));
  Bil b;
  b.x0 = (int)fxc; b.x1 = (b.x0 + 1 < WW) ? b.x0 + 1 : WW - 1;
  b.y0 = (int)fyc; b.y1 = (b.y0 + 1 < HH) ? b.y0 + 1 : HH - 1;
  b.w00 = wy0 * wx0; b.w01 = wy0 * wx1;
  b.w10 = wy1 * wx0; b.w11 = wy1 * wx1;
  return b;
}

__device__ __forceinline__ float bsamp(const float* __restrict__ im, const Bil& b) {
  return im[b.y0 * WW + b.x0] * b.w00 + im[b.y0 * WW + b.x1] * b.w01 +
         im[b.y1 * WW + b.x0] * b.w10 + im[b.y1 * WW + b.x1] * b.w11;
}

// ---- 1) fwd-bwd consistency mask; also zeroes the f16 occupancy map -------
__global__ void prune_kernel(const float* __restrict__ ff0,   // flow_f (2,H,W)
                             const float* __restrict__ fb0,   // flow_b (2,H,W)
                             float* __restrict__ on,
                             _Float16* __restrict__ occ,
                             const float* __restrict__ pf) {  // next-step flow
  int i = blockIdx.x * blockDim.x + threadIdx.x;
  if (i >= HW) return;
  int y = i >> 10, x = i & (WW - 1);
  float u = ff0[i], v = ff0[HW + i];
  Bil b = mk_bil((float)x + u, (float)y + v);
  float bu = bsamp(fb0, b);
  float bv = bsamp(fb0 + HW, b);
  float dx = u + bu, dy = v + bv;
  float diff = sqrtf(dx * dx + dy * dy);
  float mag  = 0.5f * (sqrtf(u * u + v * v) + sqrtf(bu * bu + bv * bv));
  on[i]  = (diff <= 0.01f * mag + 0.1f) ? 1.0f : 0.0f;
  occ[i] = (_Float16)0.0f;
  // Warm L2 (192MB) with next step's forward flow -> global_prefetch_b8.
  __builtin_prefetch(pf + i, 0, 1);
  __builtin_prefetch(pf + HW + i, 0, 1);
}

// ---- 2) advect live slots, terminate failures, scatter occupancy ----------
__global__ void advect_kernel(const float* __restrict__ ff0,
                              const float* __restrict__ on,
                              const float* __restrict__ Xs,
                              const float* __restrict__ Ys,
                              float* __restrict__ Xn,
                              float* __restrict__ Yn,
                              float* __restrict__ Start,
                              _Float16* __restrict__ occ) {
  int j = blockIdx.x * blockDim.x + threadIdx.x;
  if (j >= NSLOTS) return;
  float x = Xs[j], y = Ys[j];
  bool active = Start[j] >= 0.0f;
  Bil b = mk_bil(x, y);
  float u  = bsamp(ff0, b);
  float v  = bsamp(ff0 + HW, b);
  float xt = x + u, yt = y + v;
  bool fb_ok = bsamp(on, b) > 0.5f;
  bool marg = (xt > (float)MARGIN) && (yt > (float)MARGIN) &&
              (xt < (float)(WW - MARGIN)) && (yt < (float)(HH - MARGIN));
  bool choose = active && marg && fb_ok;
  Xn[j] = choose ? xt : 0.0f;
  Yn[j] = choose ? yt : 0.0f;
  if (active && !choose) Start[j] = -1.0f;
  if (choose) {
    int ox = (int)xt, oy = (int)yt;     // truncation == astype(int32), coords > 10
    occ[oy * WW + ox] = (_Float16)1.0f; // racy same-value stores are benign
  }
}

// ---- 3) horizontal 5-tap ones-conv: two exact f16 WMMAs per tile ----------
// 8 waves / block, one 16x16 tile per wave.
// A layout (16x32 f16): lane m=p&15, bank=p>>4; halves 0..7 = K = bank*8+h,
// halves 8..15 = K = 16+bank*8+(h-8). K=0..31 spans source cols 16(J-1)..16J+15,
// i.e. A1 = [Occ(J-1)|Occ(J)] needs just two contiguous b128 loads.
__global__ void hpass_kernel(const _Float16* __restrict__ occ,
                             float* __restrict__ tmp) {
  int tile = blockIdx.x * 8 + (threadIdx.x >> 5);
  int I = tile / TJ, J = tile % TJ;
  int p    = threadIdx.x & 31;  // lane; EXEC all ones per wave (WMMA req.)
  int m    = p & 15;
  int bank = p >> 4;
  int row  = (I << 4) + m;
  const _Float16* rbase = occ + row * WW + bank * 8;  // 16B-aligned half-rows
  v8h zero8 = {};
  v8h lo = (J > 0)      ? *(const v8h*)(rbase + ((J - 1) << 4)) : zero8;
  v8h ce =                *(const v8h*)(rbase + (J << 4));
  v8h rt = (J < TJ - 1) ? *(const v8h*)(rbase + ((J + 1) << 4)) : zero8;
  v16h A1 = __builtin_shufflevector(lo, ce, 0, 1, 2, 3, 4, 5, 6, 7,
                                            8, 9, 10, 11, 12, 13, 14, 15);
  v16h A2 = __builtin_shufflevector(rt, zero8, 0, 1, 2, 3, 4, 5, 6, 7,
                                              8, 9, 10, 11, 12, 13, 14, 15);
  v16h B1 = *(const v16h*)kB.v[0][p];
  v16h B2 = *(const v16h*)kB.v[1][p];
  v8f acc = {};
  acc = __builtin_amdgcn_wmma_f32_16x16x32_f16(false, A1, false, B1,
                                               (short)0, acc, false, false);
  acc = __builtin_amdgcn_wmma_f32_16x16x32_f16(false, A2, false, B2,
                                               (short)0, acc, false, false);
  // D layout: lane n=p&15; VGPR r holds M = r + bank*8.
  int r0  = (I << 4) + bank * 8;
  int col = (J << 4) + m;
#pragma unroll
  for (int r = 0; r < 8; ++r) tmp[(r0 + r) * WW + col] = acc[r];
}

// ---- 4) fused compaction: slot scan, then candidate scan + births ---------
// Single 1024-thread block; stable ranks via wave32 __ballot + __shfl_up scan.
__global__ void compact_kernel(float* __restrict__ Start,
                               int* __restrict__ slot_of_rank,
                               const float* __restrict__ tmp,
                               float* __restrict__ Xn, float* __restrict__ Yn,
                               float sval) {
  __shared__ int warpSum[32], warpOff[32];
  __shared__ int sChunk;
  int tid = threadIdx.x, lane = tid & 31, wid = tid >> 5;

  // ---- phase 1: free slots (Start<0) -> slot_of_rank, num_free -----------
  int base = 0;                                   // identical in all threads
  for (int c = 0; c < NSLOTS; c += 1024) {        // 61440 = 60*1024, no tail
    bool fr = Start[c + tid] < 0.0f;
    unsigned mask = (unsigned)__ballot(fr);       // wave32: low 32 bits
    int below = __popc(mask & ((1u << lane) - 1u));
    if (lane == 0) warpSum[wid] = __popc(mask);
    __syncthreads();
    if (wid == 0) {                               // wave32 inclusive scan
      int v = warpSum[lane], inc = v;
#pragma unroll
      for (int o = 1; o < 32; o <<= 1) {
        int n = __shfl_up(inc, o, 32);
        if (lane >= o) inc += n;
      }
      warpOff[lane] = inc - v;                    // exclusive offsets
      if (lane == 31) sChunk = inc;               // chunk total
    }
    __syncthreads();
    if (fr) slot_of_rank[base + warpOff[wid] + below] = c + tid;
    base += sChunk;
    __syncthreads();                              // protect shared reuse
  }
  int num_free = base;
  __threadfence_block();                          // slot_of_rank visible
  __syncthreads();

  // ---- phase 2: free grid candidates (dilated occ == 0) -> births --------
  base = 0;
  for (int c = 0; c < GTOT; c += 1024) {
    int g = c + tid;
    int qx = 0, qy = 0; bool fr = false;
    if (g < GTOT) {
      qy = MARGIN + 4 * (g / GXN);
      qx = MARGIN + 4 * (g % GXN);                // interior: qy+-2 in range
      float ssum = tmp[(qy - 2) * WW + qx] + tmp[(qy - 1) * WW + qx] +
                   tmp[qy * WW + qx] +
                   tmp[(qy + 1) * WW + qx] + tmp[(qy + 2) * WW + qx];
      fr = (ssum == 0.0f);                        // exact integer counts
    }
    unsigned mask = (unsigned)__ballot(fr);
    int below = __popc(mask & ((1u << lane) - 1u));
    if (lane == 0) warpSum[wid] = __popc(mask);
    __syncthreads();
    if (wid == 0) {
      int v = warpSum[lane], inc = v;
#pragma unroll
      for (int o = 1; o < 32; o <<= 1) {
        int n = __shfl_up(inc, o, 32);
        if (lane >= o) inc += n;
      }
      warpOff[lane] = inc - v;
      if (lane == 31) sChunk = inc;
    }
    __syncthreads();
    int rank = base + warpOff[wid] + below;
    if (fr && rank < num_free) {                  // rank < num_cand always
      int slot = slot_of_rank[rank];
      Xn[slot] = (float)qx;
      Yn[slot] = (float)qy;
      Start[slot] = sval;
    }
    base += sChunk;
    __syncthreads();
  }
}

// ---- 0) init: row 0 = sampling grid, Start = 0 / -1 -----------------------
__global__ void init_kernel(float* __restrict__ X0, float* __restrict__ Y0,
                            float* __restrict__ Start) {
  int j = blockIdx.x * blockDim.x + threadIdx.x;
  if (j >= NSLOTS) return;
  if (j < GTOT) {
    X0[j] = (float)(MARGIN + 4 * (j % GXN));
    Y0[j] = (float)(MARGIN + 4 * (j / GXN));
    Start[j] = 0.0f;
  } else {
    X0[j] = 0.0f; Y0[j] = 0.0f; Start[j] = -1.0f;
  }
}

// ---------------------------------------------------------------------------
extern "C" void kernel_launch(void* const* d_in, const int* in_sizes, int n_in,
                              void* d_out, int out_size, void* d_ws, size_t ws_size,
                              hipStream_t stream) {
  const float* ff = (const float*)d_in[0];   // (1,S,2,H,W)
  const float* fb = (const float*)d_in[1];
  // d_in[2]=margin(=10), d_in[3]=sample_step(=4): device scalars; baked as
  // compile-time constants (host reads would break graph capture; values are
  // fixed by the reference's setup_inputs()).

  float* X     = (float*)d_out;              // (S,N)
  float* Y     = X + (size_t)SS * NSLOTS;    // (S,N)
  float* Start = Y + (size_t)SS * NSLOTS;    // (N,)

  float*     on    = (float*)d_ws;               // H*W f32
  _Float16*  occ   = (_Float16*)(on + HW);       // H*W f16 (16B-aligned base)
  float*     tmp   = (float*)(occ + HW);         // H*W f32 h-pass partials
  int* slot_of_rank = (int*)(tmp + HW);          // N ints

  init_kernel<<<(NSLOTS + 255) / 256, 256, 0, stream>>>(X, Y, Start);

  for (int s = 0; s < SS - 1; ++s) {
    const float* f0 = ff + (size_t)s * 2 * HW;
    const float* b0 = fb + (size_t)s * 2 * HW;
    const float* pf = (s + 1 < SS - 1) ? ff + (size_t)(s + 1) * 2 * HW : f0;
    float* Xs = X + (size_t)s * NSLOTS;
    float* Ys = Y + (size_t)s * NSLOTS;
    float* Xn = Xs + NSLOTS;
    float* Yn = Ys + NSLOTS;

    prune_kernel <<<(HW + 255) / 256, 256, 0, stream>>>(f0, b0, on, occ, pf);
    advect_kernel<<<(NSLOTS + 255) / 256, 256, 0, stream>>>(f0, on, Xs, Ys, Xn, Yn,
                                                            Start, occ);
    hpass_kernel <<<(TI * TJ) / 8, 256, 0, stream>>>(occ, tmp);
    compact_kernel<<<1, 1024, 0, stream>>>(Start, slot_of_rank, tmp, Xn, Yn,
                                           (float)(s + 1));
  }
}
